// sentence_encoder_67207648248115
// MI455X (gfx1250) — compile-verified, matrix operand order checked by reading, not run on hardware
//
#include <hip/hip_runtime.h>
#include <hip/hip_bf16.h>
#include <math.h>

#define BB 4096
#define TT 16
#define EE 256
#define HH 512
#define NG 2048   // 4*HH

typedef __attribute__((ext_vector_type(16))) __bf16 v16bf;
typedef __attribute__((ext_vector_type(8)))  float  v8f;

union Frag { unsigned int u[8]; v16bf v; };

__device__ __forceinline__ unsigned short f2bf(float f) {
  unsigned int u = __float_as_uint(f);
  u += 0x7FFFu + ((u >> 16) & 1u);      // round-to-nearest-even
  return (unsigned short)(u >> 16);
}

__device__ __forceinline__ float sigm(float x) {
  return 1.0f / (1.0f + __expf(-x));
}

// K-pair offset inside a 32-wide K chunk for 16-bit A/B fragments
// (ISA 7.12.2: VGPR v holds K pairs, +8 for the upper half-wave).
__device__ __forceinline__ int kpair(int v, int half) {
  return (half << 3) + ((v < 4) ? (v << 1) : (16 + ((v - 4) << 1)));
}

// ---------------- embedding gather: emb_table[tokens] -> bf16 [T][B][E] ----
__global__ __launch_bounds__(256) void gather_emb_kernel(
    const int* __restrict__ tokens, const float* __restrict__ emb,
    unsigned short* __restrict__ xbf) {
  int bt = blockIdx.x;            // b*T + t
  int b = bt >> 4, t = bt & 15;
  int e = threadIdx.x;            // 0..255
  int tok = tokens[b * TT + t];
  float v = emb[(size_t)tok * EE + e];
  xbf[((size_t)t * BB + b) * EE + e] = f2bf(v);
}

// -------- weight prep: [W_ih | W_hh] fp32 -> bf16 row-major [NG][KX+HH] ----
__global__ __launch_bounds__(256) void wcat_kernel(
    const float* __restrict__ Wih, const float* __restrict__ Whh,
    unsigned short* __restrict__ Wc, int KX) {
  int K = KX + HH;
  size_t idx = (size_t)blockIdx.x * 256 + threadIdx.x;
  size_t total = (size_t)NG * K;
  if (idx >= total) return;
  int n = (int)(idx / K), k = (int)(idx % K);
  float v = (k < KX) ? Wih[(size_t)n * KX + k] : Whh[(size_t)n * HH + (k - KX)];
  Wc[idx] = f2bf(v);
}

// --------------------------- recurrent LSTM layer --------------------------
// Workgroup = 16 batch rows, 16 waves. Wave w owns gate columns
// j in [32w, 32w+32) for all four gates; c/h state lives in VGPRs across t.
template<int KX, bool LAST>
__global__ __launch_bounds__(512) void lstm_layer_kernel(
    const unsigned short* __restrict__ xin,    // bf16 [T][B][KX]
    const unsigned short* __restrict__ Wcat,   // bf16 [NG][KX+HH]
    const float* __restrict__ bias,            // [NG]
    const int*  __restrict__ lengths,          // [B]
    const float* __restrict__ h0,              // [HH]
    const float* __restrict__ c0,              // [HH]
    unsigned short* __restrict__ hout_bf,      // bf16 [T][B][HH]  (layer 0)
    float* __restrict__ out,                   // f32 [B][T][HH]   (layer 1)
    float* __restrict__ hn_out) {              // f32 [B][HH]      (layer 1)
  constexpr int K   = KX + HH;
  constexpr int KCX = KX / 32;                 // x-sweep chunks
  constexpr int KCH = HH / 32;                 // h-sweep chunks
  const int tid  = threadIdx.x;
  const int wave = tid >> 5, lane = tid & 31;
  const int half = lane >> 4, lq = lane & 15;
  const int row0 = blockIdx.x * 16;

  __shared__ unsigned short hbf[16 * HH];      // published h (bf16), 16 KB

  float hreg[16], creg[16];                    // state owned by this wave
  #pragma unroll
  for (int jti = 0; jti < 2; ++jti) {
    int j = (2 * wave + jti) * 16 + lq;
    #pragma unroll
    for (int v = 0; v < 8; ++v) {
      hreg[jti * 8 + v] = h0[j];
      creg[jti * 8 + v] = c0[j];
    }
  }
  int lenM[8];
  #pragma unroll
  for (int v = 0; v < 8; ++v) lenM[v] = lengths[row0 + v + 8 * half];

  for (int idx = tid; idx < 16 * HH; idx += 512)
    hbf[idx] = f2bf(h0[idx & (HH - 1)]);
  __syncthreads();

  // t-invariant bases
  const unsigned short* wbase[8];
  #pragma unroll
  for (int gb = 0; gb < 4; ++gb)
    #pragma unroll
    for (int jti = 0; jti < 2; ++jti) {
      int n = gb * HH + (2 * wave + jti) * 16 + lq;   // weight row = gate col
      wbase[gb * 2 + jti] = Wcat + (size_t)n * K;
    }
  float bv2[8];
  #pragma unroll
  for (int gb = 0; gb < 4; ++gb)
    #pragma unroll
    for (int jti = 0; jti < 2; ++jti)
      bv2[gb * 2 + jti] = bias[gb * HH + (2 * wave + jti) * 16 + lq];

  for (int t = 0; t < TT; ++t) {
    v8f acc[8];                                // [gate 0..3][jtile 0..1]
    #pragma unroll
    for (int i = 0; i < 8; ++i)
      #pragma unroll
      for (int v = 0; v < 8; ++v) acc[i][v] = bv2[i];

    // ---- sweep 1: x contribution (A from global, compile-time trip count)
    const unsigned short* xrow = xin + ((size_t)t * BB + row0 + lq) * KX;
    #pragma unroll 2
    for (int kc = 0; kc < KCX; ++kc) {
      const int kbase = kc * 32;
      Frag fa;
      #pragma unroll
      for (int v = 0; v < 8; ++v)
        fa.u[v] = *(const unsigned int*)(xrow + kbase + kpair(v, half));
      v16bf a = fa.v;
      #pragma unroll
      for (int i = 0; i < 8; ++i) {
        Frag fb;
        #pragma unroll
        for (int v = 0; v < 8; ++v)
          fb.u[v] = *(const unsigned int*)(wbase[i] + kbase + kpair(v, half));
        acc[i] = __builtin_amdgcn_wmma_f32_16x16x32_bf16(
            false, a, false, fb.v, (short)0, acc[i], false, false);
      }
    }

    // ---- sweep 2: recurrent h contribution (A from LDS)
    const unsigned short* hrow = &hbf[lq * HH];
    #pragma unroll 2
    for (int kc = 0; kc < KCH; ++kc) {
      const int kbase = kc * 32;
      Frag fa;
      #pragma unroll
      for (int v = 0; v < 8; ++v)
        fa.u[v] = *(const unsigned int*)(hrow + kbase + kpair(v, half));
      v16bf a = fa.v;
      #pragma unroll
      for (int i = 0; i < 8; ++i) {
        Frag fb;
        #pragma unroll
        for (int v = 0; v < 8; ++v)
          fb.u[v] = *(const unsigned int*)(wbase[i] + KX + kbase + kpair(v, half));
        acc[i] = __builtin_amdgcn_wmma_f32_16x16x32_bf16(
            false, a, false, fb.v, (short)0, acc[i], false, false);
      }
    }
    __syncthreads();   // all waves finished reading hbf for this step

    // In-register cell update: i/f/g/o for the same (M,j) share (v,lane).
    #pragma unroll
    for (int jti = 0; jti < 2; ++jti)
      #pragma unroll
      for (int v = 0; v < 8; ++v) {
        int s = jti * 8 + v;
        float ig = sigm(acc[0 * 2 + jti][v]);
        float fg = sigm(acc[1 * 2 + jti][v]);
        float gg = tanhf(acc[2 * 2 + jti][v]);
        float og = sigm(acc[3 * 2 + jti][v]);
        float cn = fg * creg[s] + ig * gg;
        float hv = og * tanhf(cn);
        bool m = (t < lenM[v]);
        if (m) { creg[s] = cn; hreg[s] = hv; }
        int M = v + 8 * half;
        int j = (2 * wave + jti) * 16 + lq;
        unsigned short hb = f2bf(hreg[s]);
        hbf[M * HH + j] = hb;                           // publish for next step
        if (!LAST) {
          hout_bf[((size_t)t * BB + row0 + M) * HH + j] = hb;
        } else {
          out[((size_t)(row0 + M) * TT + t) * HH + j] = m ? hv : 0.0f;
        }
      }
    __syncthreads();   // hbf fully updated before next step's GEMM
  }

  if (LAST) {
    #pragma unroll
    for (int jti = 0; jti < 2; ++jti)
      #pragma unroll
      for (int v = 0; v < 8; ++v) {
        int M = v + 8 * half;
        int j = (2 * wave + jti) * 16 + lq;
        hn_out[(size_t)(row0 + M) * HH + j] = hreg[jti * 8 + v];
      }
  }
}

extern "C" void kernel_launch(void* const* d_in, const int* in_sizes, int n_in,
                              void* d_out, int out_size, void* d_ws, size_t ws_size,
                              hipStream_t stream) {
  (void)in_sizes; (void)n_in; (void)out_size; (void)ws_size;
  const int*   tokens  = (const int*)d_in[0];
  const int*   lengths = (const int*)d_in[1];
  const float* emb     = (const float*)d_in[2];
  const float* Wih0    = (const float*)d_in[3];
  const float* Whh0    = (const float*)d_in[4];
  const float* b0      = (const float*)d_in[5];
  const float* Wih1    = (const float*)d_in[6];
  const float* Whh1    = (const float*)d_in[7];
  const float* b1      = (const float*)d_in[8];
  const float* h0      = (const float*)d_in[9];
  const float* c0      = (const float*)d_in[10];

  char* ws = (char*)d_ws;
  size_t off = 0;
  unsigned short* xbf = (unsigned short*)(ws + off);   off += (size_t)TT * BB * EE * 2; // 32 MB
  unsigned short* hs0 = (unsigned short*)(ws + off);   off += (size_t)TT * BB * HH * 2; // 64 MB
  unsigned short* Wc0 = (unsigned short*)(ws + off);   off += (size_t)NG * (EE + HH) * 2; // 3 MB
  unsigned short* Wc1 = (unsigned short*)(ws + off);

  float* out = (float*)d_out;                 // [B][T][H]
  float* hn  = out + (size_t)BB * TT * HH;    // [B][H]

  gather_emb_kernel<<<BB * TT, 256, 0, stream>>>(tokens, emb, xbf);
  wcat_kernel<<<(NG * (EE + HH) + 255) / 256, 256, 0, stream>>>(Wih0, Whh0, Wc0, EE);
  wcat_kernel<<<(NG * (HH + HH) + 255) / 256, 256, 0, stream>>>(Wih1, Whh1, Wc1, HH);

  lstm_layer_kernel<EE, false><<<BB / 16, 512, 0, stream>>>(
      xbf, Wc0, b0, lengths, h0, c0, hs0, nullptr, nullptr);
  lstm_layer_kernel<HH, true><<<BB / 16, 512, 0, stream>>>(
      hs0, Wc1, b1, lengths, h0 + HH, c0 + HH, nullptr, out, hn);
}
